// FactorModel_90847148245270
// MI455X (gfx1250) — matrix operand order repaired and checked by reading, not compiled, not run on hardware
//
#include <hip/hip_runtime.h>
#include <hip/hip_bf16.h>

// ---------------------------------------------------------------------------
// FactorModel on MI455X (gfx1250, wave32, WMMA).
//
// Structure exploit: the edge list is the complete graph within each of the
// B=128 blocks, so every GCN layer is dense:
//     y[n] = (sum_block(h@W) - (h@W)[n]) / 127 + b
// Pipeline:
//   pack_kernel  : repack We0/We1/We2 (f32 -> bf16) into WMMA B-fragment order
//   gcn_kernel   : 3 dense "GCN" layers, scalar f32; emits h (output 0) and
//                  xb (f32 + bf16 copies) into workspace
//   node_kernel  : node MLP (64->64->64->10), scalar f32; writes the 128
//                  diagonal 4x4 blocks of U
//   edge_kernel  : edge MLP (128->128->128->16) with v_wmma_f32_16x16x32_bf16,
//                  f32 accumulate; scatters every off-diagonal 4x4 block of U
// ---------------------------------------------------------------------------

typedef __bf16 bf16_t;
typedef __attribute__((ext_vector_type(16))) __bf16 v16bf;
typedef __attribute__((ext_vector_type(8)))  float  v8f;

#define BB   128
#define NN   128
#define DIN  6
#define HH   32
#define DMP  64
#define EUD  8128
#define TPB  508            // edge tiles (of 16 edges) per block: 8128/16
#define NTOT (BB*NN)
#define SLOPE 0.1f
#define INV127 (1.0f/127.0f)

union ABfrag { v16bf v; uint4 q[2]; };

// ---------------------------------------------------------------------------
// Weight packer: B-fragment order for v_wmma_*_16x16x32_bf16.
// Tile (kt,nt) covers K rows [kt*32, kt*32+32), N cols [nt*16, nt*16+16).
// Lane<16 holds K = kt*32 + {0..7, 16..23} for col nt*16+lane;
// lane>=16 holds K = kt*32 + {8..15, 24..31} for col nt*16+(lane-16).
// Packed element index = ((kt*n16tiles + nt)*32 + lane)*16 + i.
// ---------------------------------------------------------------------------
__global__ __launch_bounds__(256) void pack_kernel(
    const float* __restrict__ W, bf16_t* __restrict__ out,
    int n16tiles, int ncols, int total)
{
  int idx = (int)blockIdx.x * 256 + (int)threadIdx.x;
  if (idx >= total) return;
  int i    = idx & 15;
  int lane = (idx >> 4) & 31;
  int tile = idx >> 9;
  int nt   = tile % n16tiles;
  int kt   = tile / n16tiles;
  int half = lane >> 4;
  int krow = kt*32 + half*8 + (i & 7) + ((i >> 3) << 4);
  int col  = nt*16 + (lane & 15);
  out[idx] = (bf16_t)W[krow * ncols + col];
}

// ---------------------------------------------------------------------------
// GCN combine: given per-node y[c] (c < C), compute
//   out[c] = lrelu?((colsum(y) - y[c]) * (1/127) + bias[c])
// using an LDS column reduction across the 128 nodes of the block.
// ---------------------------------------------------------------------------
__device__ __forceinline__ void gcn_combine(
    float* buf /*128*64*/, float* part /*4*64*/,
    const float* y, float* out, int C,
    const float* __restrict__ bias, bool relu, int n)
{
  for (int c = 0; c < C; ++c) buf[n*64 + c] = y[c];
  __syncthreads();
  if (C == 32) {
    int seg = n >> 5, col = n & 31;
    float p = 0.f;
    for (int r = 0; r < 32; ++r) p += buf[(seg*32 + r)*64 + col];
    part[seg*64 + col] = p;
  } else { // C == 64
    int seg = n >> 6, col = n & 63;
    float p = 0.f;
    for (int r = 0; r < 64; ++r) p += buf[(seg*64 + r)*64 + col];
    part[seg*64 + col] = p;
  }
  __syncthreads();
  for (int c = 0; c < C; ++c) {
    float S = (C == 32)
      ? (part[c] + part[64+c] + part[128+c] + part[192+c])
      : (part[c] + part[64+c]);
    float h = (S - y[c]) * INV127 + bias[c];
    if (relu) h = (h >= 0.f) ? h : SLOPE * h;
    out[c] = h;
  }
  __syncthreads();
}

__global__ __launch_bounds__(128) void gcn_kernel(
    const float* __restrict__ x,
    const float* __restrict__ Wg0, const float* __restrict__ bg0,
    const float* __restrict__ Wg1, const float* __restrict__ bg1,
    const float* __restrict__ Wg2, const float* __restrict__ bg2,
    float* __restrict__ h_out, float* __restrict__ xb_f, bf16_t* __restrict__ xb_b)
{
  __shared__ float buf[NN*64];
  __shared__ float part[4*64];
  const int n = (int)threadIdx.x;
  const int b = (int)blockIdx.x;
  const int node = b*NN + n;

  float cur[64], nxt[64];
  #pragma unroll
  for (int d = 0; d < DIN; ++d) cur[d] = x[(size_t)node*DIN + d];

  // layer 0: 6 -> 32
  for (int c = 0; c < HH; ++c) {
    float acc = 0.f;
    #pragma unroll
    for (int d = 0; d < DIN; ++d) acc += cur[d] * Wg0[d*HH + c];
    nxt[c] = acc;
  }
  gcn_combine(buf, part, nxt, cur, HH, bg0, true, n);

  // layer 1: 32 -> 32
  for (int c = 0; c < HH; ++c) {
    float acc = 0.f;
    for (int d = 0; d < HH; ++d) acc += cur[d] * Wg1[d*HH + c];
    nxt[c] = acc;
  }
  gcn_combine(buf, part, nxt, cur, HH, bg1, true, n);

  // layer 2: 32 -> 64  (no lrelu on final GCN layer)
  for (int c = 0; c < DMP; ++c) {
    float acc = 0.f;
    for (int d = 0; d < HH; ++d) acc += cur[d] * Wg2[d*DMP + c];
    nxt[c] = acc;
  }
  gcn_combine(buf, part, nxt, cur, DMP, bg2, false, n);

  #pragma unroll
  for (int c = 0; c < DMP; ++c) {
    float v = cur[c];
    h_out[(size_t)node*DMP + c] = v;
    xb_f [(size_t)node*DMP + c] = v;
    xb_b [(size_t)node*DMP + c] = (bf16_t)v;
  }
}

// ---------------------------------------------------------------------------
// Node MLP + U diagonal blocks. One thread per node, weights staged in LDS.
// ---------------------------------------------------------------------------
__global__ __launch_bounds__(128) void node_kernel(
    const float* __restrict__ xb,
    const float* __restrict__ Wn0, const float* __restrict__ bn0,
    const float* __restrict__ Wn1, const float* __restrict__ bn1,
    const float* __restrict__ Wn2, const float* __restrict__ bn2,
    float* __restrict__ U)
{
  __shared__ float w[DMP*DMP];
  const int n = (int)threadIdx.x;
  const int b = (int)blockIdx.x;
  const int node = b*NN + n;

  float cur[64], nxt[64];
  for (int c = 0; c < DMP; ++c) cur[c] = xb[(size_t)node*DMP + c];

  for (int i = n; i < DMP*DMP; i += NN) w[i] = Wn0[i];
  __syncthreads();
  for (int c = 0; c < DMP; ++c) {
    float acc = bn0[c];
    for (int d = 0; d < DMP; ++d) acc += cur[d] * w[d*DMP + c];
    nxt[c] = (acc >= 0.f) ? acc : SLOPE * acc;
  }
  __syncthreads();

  for (int i = n; i < DMP*DMP; i += NN) w[i] = Wn1[i];
  __syncthreads();
  for (int c = 0; c < DMP; ++c) {
    float acc = bn1[c];
    for (int d = 0; d < DMP; ++d) acc += nxt[d] * w[d*DMP + c];
    cur[c] = (acc >= 0.f) ? acc : SLOPE * acc;
  }
  __syncthreads();

  for (int i = n; i < DMP*10; i += NN) w[i] = Wn2[i];
  __syncthreads();
  float o[10];
  for (int c = 0; c < 10; ++c) {
    float acc = bn2[c];
    for (int d = 0; d < DMP; ++d) acc += cur[d] * w[d*10 + c];
    o[c] = acc;
  }

  const int smat[16] = {0,1,2,3, 1,4,5,6, 2,5,7,8, 3,6,8,9};
  float* Ub = U + (size_t)b * (4*NN) * (4*NN);
  #pragma unroll
  for (int s = 0; s < 16; ++s)
    Ub[(size_t)(4*n + (s >> 2)) * (4*NN) + 4*n + (s & 3)] = o[smat[s]];
}

// ---------------------------------------------------------------------------
// Edge MLP: one wave32 per 16-edge tile; bf16 WMMA, f32 accumulate.
// ---------------------------------------------------------------------------
__global__ __launch_bounds__(256) void edge_kernel(
    const bf16_t* __restrict__ xb, const int* __restrict__ ud,
    const bf16_t* __restrict__ wp0, const bf16_t* __restrict__ wp1,
    const bf16_t* __restrict__ wp2,
    const float* __restrict__ be0, const float* __restrict__ be1,
    const float* __restrict__ be2, float* __restrict__ U)
{
  __shared__ __align__(16) bf16_t slds[8][16*128];   // 4 KB scratch per wave
  const int wave = (int)threadIdx.x >> 5;
  const int lane = (int)threadIdx.x & 31;
  const int tile = (int)blockIdx.x * 8 + wave;       // < 128*508 exactly
  const int b    = tile / TPB;
  const int e0   = (tile - b*TPB) * 16;
  const int half = lane >> 4;
  const int lm   = lane & 15;
  const int Mb   = half * 8;
  bf16_t* myl = slds[wave];

  // A-fragment row for this lane: row M = lm -> edge (e0+lm) = (ni, nj).
  const int ni = ud[(e0 + lm)*2 + 0];
  const int nj = ud[(e0 + lm)*2 + 1];
  const bf16_t* rowi = xb + (size_t)(b*NN + ni) * DMP;
  const bf16_t* rowj = xb + (size_t)(b*NN + nj) * DMP;

  ABfrag a[4];
  // Layer-0 A fragments straight from global xb_bf16. Lane layout per ISA:
  // lane<16: K = kt*32 + {0..7, 16..23}; lane>=16: K = kt*32 + {8..15, 24..31}.
  // Features 0..63 come from node i, 64..127 from node j; runs of 8 (16B,
  // aligned) never straddle the boundary.
  #pragma unroll
  for (int kt = 0; kt < 4; ++kt) {
    int f0 = kt*32 + half*8;
    int f1 = f0 + 16;
    const bf16_t* p0 = (f0 < DMP) ? (rowi + f0) : (rowj + (f0 - DMP));
    const bf16_t* p1 = (f1 < DMP) ? (rowi + f1) : (rowj + (f1 - DMP));
    a[kt].q[0] = *(const uint4*)p0;
    a[kt].q[1] = *(const uint4*)p1;
  }

  // ---- layers 0 and 1: 128 -> 128, lrelu, via LDS transpose -------------
  const bf16_t* wps[2] = { wp0, wp1 };
  const float*  bes[2] = { be0, be1 };
  for (int layer = 0; layer < 2; ++layer) {
    const v16bf* wv = (const v16bf*)wps[layer];
    const float* bias = bes[layer];
    #pragma unroll
    for (int nt = 0; nt < 8; ++nt) {
      float bv = bias[nt*16 + lm];               // C[m][n] init = bias[n]
      v8f acc = { bv, bv, bv, bv, bv, bv, bv, bv };
      #pragma unroll
      for (int kt = 0; kt < 4; ++kt) {
        v16bf bmat = wv[(kt*8 + nt)*32 + lane];  // 32B contiguous per lane
        acc = __builtin_amdgcn_wmma_f32_16x16x32_bf16(
                  false, a[kt].v, false, bmat, (short)0, acc, false, false);
      }
      #pragma unroll
      for (int r = 0; r < 8; ++r) {              // C: lane holds col n=lm, row Mb+r
        float xv = acc[r];
        xv = (xv >= 0.f) ? xv : SLOPE * xv;
        myl[(Mb + r)*128 + nt*16 + lm] = (bf16_t)xv;
      }
    }
    asm volatile("s_wait_dscnt 0x0" ::: "memory"); // cross-lane LDS handoff
    #pragma unroll
    for (int kt = 0; kt < 4; ++kt) {             // reload A-frag from LDS
      int f0 = kt*32 + half*8;
      a[kt].q[0] = *(const uint4*)&myl[lm*128 + f0];
      a[kt].q[1] = *(const uint4*)&myl[lm*128 + f0 + 16];
    }
  }

  // ---- layer 2: 128 -> 16, no activation --------------------------------
  {
    const v16bf* wv = (const v16bf*)wp2;
    float bv = be2[lm];
    v8f acc = { bv, bv, bv, bv, bv, bv, bv, bv };
    #pragma unroll
    for (int kt = 0; kt < 4; ++kt) {
      v16bf bmat = wv[kt*32 + lane];
      acc = __builtin_amdgcn_wmma_f32_16x16x32_bf16(
                false, a[kt].v, false, bmat, (short)0, acc, false, false);
    }
    float* ef = (float*)myl;                     // reuse scratch as 16x16 f32
    #pragma unroll
    for (int r = 0; r < 8; ++r) ef[(Mb + r)*16 + lm] = acc[r];
  }
  asm volatile("s_wait_dscnt 0x0" ::: "memory");

  // ---- scatter 4x4 blocks into U at (i,j) and (j,i) ----------------------
  {
    const float* ef = (const float*)myl;
    float* Ub = U + (size_t)b * (4*NN) * (4*NN);
    #pragma unroll
    for (int q = 0; q < 4; ++q) {
      int task = lane + 32*q;                    // 16 edges * 2 syms * 4 rows
      int m   = task >> 3;
      int sym = (task >> 2) & 1;
      int r   = task & 3;
      float4 row = *(const float4*)&ef[m*16 + 4*r];
      int i2 = ud[(e0 + m)*2 + 0];
      int j2 = ud[(e0 + m)*2 + 1];
      if (sym) { int t = i2; i2 = j2; j2 = t; }
      *(float4*)&Ub[(size_t)(4*i2 + r) * (4*NN) + 4*j2] = row;
    }
  }
}

// ---------------------------------------------------------------------------
extern "C" void kernel_launch(void* const* d_in, const int* in_sizes, int n_in,
                              void* d_out, int out_size, void* d_ws, size_t ws_size,
                              hipStream_t stream) {
  (void)in_sizes; (void)n_in; (void)out_size; (void)ws_size;
  const float* x   = (const float*)d_in[0];
  // d_in[1] edge_index, d_in[2] edge_map: unused (complete-graph structure).
  const int*   ud  = (const int*)  d_in[3];
  const float* Wg0 = (const float*)d_in[4];  const float* bg0 = (const float*)d_in[5];
  const float* Wg1 = (const float*)d_in[6];  const float* bg1 = (const float*)d_in[7];
  const float* Wg2 = (const float*)d_in[8];  const float* bg2 = (const float*)d_in[9];
  const float* Wn0 = (const float*)d_in[10]; const float* bn0 = (const float*)d_in[11];
  const float* Wn1 = (const float*)d_in[12]; const float* bn1 = (const float*)d_in[13];
  const float* Wn2 = (const float*)d_in[14]; const float* bn2 = (const float*)d_in[15];
  const float* We0 = (const float*)d_in[16]; const float* be0 = (const float*)d_in[17];
  const float* We1 = (const float*)d_in[18]; const float* be1 = (const float*)d_in[19];
  const float* We2 = (const float*)d_in[20]; const float* be2 = (const float*)d_in[21];

  float* h_out = (float*)d_out;
  float* U     = (float*)d_out + (size_t)NTOT * DMP;   // outputs concat: h then U

  // Workspace: xb f32 (4 MB) | xb bf16 (2 MB) | packed We0/We1/We2 (~68 KB)
  char*   ws   = (char*)d_ws;
  float*  xb_f = (float*) ws;
  bf16_t* xb_b = (bf16_t*)(ws + (size_t)4*1024*1024);
  bf16_t* wp0  = (bf16_t*)(ws + (size_t)6*1024*1024);
  bf16_t* wp1  = wp0 + 4*8*512;
  bf16_t* wp2  = wp1 + 4*8*512;

  pack_kernel<<<64, 256, 0, stream>>>(We0, wp0, 8, 2*DMP, 4*8*512);
  pack_kernel<<<64, 256, 0, stream>>>(We1, wp1, 8, 2*DMP, 4*8*512);
  pack_kernel<<<8,  256, 0, stream>>>(We2, wp2, 1, 16,    4*1*512);

  gcn_kernel <<<BB, NN, 0, stream>>>(x, Wg0, bg0, Wg1, bg1, Wg2, bg2,
                                     h_out, xb_f, xb_b);
  node_kernel<<<BB, NN, 0, stream>>>(xb_f, Wn0, bn0, Wn1, bn1, Wn2, bn2, U);
  edge_kernel<<<(BB*TPB)/8, 256, 0, stream>>>(xb_b, ud, wp0, wp1, wp2,
                                              be0, be1, be2, U);
}
